// Attention_49907519979644
// MI455X (gfx1250) — compile-verified, matrix operand order checked by reading, not run on hardware
//
#include <hip/hip_runtime.h>
#include <hip/hip_bf16.h>

// ---------------------------------------------------------------------------
// MHA block for MI455X (gfx1250, wave32, WMMA).
// All matmuls: V_WMMA_F32_16X16X32_BF16 (bf16 operands, f32 accum).
// Fragment layouts (CDNA5 ISA 7.12.2):
//   A (16x32): lane L -> row M = L%16; elem e -> K = (e<8?0:16) + 8*(L/16) + (e&7)
//   B (32x16): lane L -> col N = L%16; same K mapping
//   C/D (16x16 f32): lane L -> col N = L%16; elem r -> row M = r + 8*(L/16)
// Tile staging uses GLOBAL_LOAD_ASYNC_TO_LDS_B128 (ASYNCcnt) double-buffered.
// ---------------------------------------------------------------------------

typedef __bf16 bf16_t;
typedef __attribute__((ext_vector_type(16))) __bf16 v16bf;
typedef __attribute__((ext_vector_type(8)))  __bf16 v8bf;
typedef __attribute__((ext_vector_type(4)))  __bf16 v4bf;
typedef __attribute__((ext_vector_type(8)))  float  v8f;

#define DIM     768
#define THREEC  2304
#define NTOK    8192      // B*N = 8*1024
#define HEADS   12
#define HDIM    64
#define SEQ     1024
#define SCALE_Q 0.125f    // 64^-0.5

#define A_STRIDE 40       // 32 k-elems + 8 pad  (80B rows: conflict-free b128 reads)
#define KV_STRIDE 72      // 64 elems + 8 pad    (144B rows: conflict-free b128 reads)

static __device__ inline v8f zero8() {
  v8f z = {0.f, 0.f, 0.f, 0.f, 0.f, 0.f, 0.f, 0.f};
  return z;
}

// 16-bf16 fragment: chunk0 = p[0..7], chunk1 = p[16..23] (global or LDS ptr).
static __device__ inline v16bf ld_frag(const bf16_t* p) {
  v8bf lo = *reinterpret_cast<const v8bf*>(p);
  v8bf hi = *reinterpret_cast<const v8bf*>(p + 16);
  v16bf r;
#pragma unroll
  for (int i = 0; i < 8; ++i) { r[i] = lo[i]; r[i + 8] = hi[i]; }
  return r;
}

static __device__ inline v8f wmma_bf16(v16bf a, v16bf b, v8f c) {
  return __builtin_amdgcn_wmma_f32_16x16x32_bf16(false, a, false, b,
                                                 (short)0, c, false, false);
}

// Per-lane async 16B copy global -> LDS (tracked by ASYNCcnt).
static __device__ inline void async_copy_b128(unsigned lds_byte_off, const bf16_t* g) {
  asm volatile("global_load_async_to_lds_b128 %0, %1, off"
               :: "v"(lds_byte_off), "v"(g) : "memory");
}
// Low 32 bits of a generic pointer to LDS == LDS byte address (aperture rule).
static __device__ inline unsigned lds_addr(const void* p) {
  return (unsigned)(unsigned long long)p;
}

// ---------------------------------------------------------------------------
// fp32 -> bf16 conversion
// ---------------------------------------------------------------------------
__global__ void cvt_f32_bf16(const float* __restrict__ s, bf16_t* __restrict__ d, int n4) {
  int i = blockIdx.x * blockDim.x + threadIdx.x;
  if (i >= n4) return;
  float4 v = reinterpret_cast<const float4*>(s)[i];
  v4bf o;
  o[0] = (bf16_t)v.x; o[1] = (bf16_t)v.y; o[2] = (bf16_t)v.z; o[3] = (bf16_t)v.w;
  *reinterpret_cast<v4bf*>(d + 4 * i) = o;
}

// ---------------------------------------------------------------------------
// QKV GEMM: qkv[t,o] = sum_c x[t,c] * w_qkv[o,c]
// Block 256 thr (8 waves, 2M x 4N). WG tile 64(M) x 256(N); wave tile 32x64.
// A tile (64x32) staged in LDS via async copy, double buffered; B from global.
// Scatter: Q*0.125 -> [B,H,N,64], K -> [B,H,N,64], V -> [B,H,64,N] transposed.
// ---------------------------------------------------------------------------
__global__ __launch_bounds__(256) void qkv_gemm_kernel(
    const bf16_t* __restrict__ Xb, const bf16_t* __restrict__ Wb,
    bf16_t* __restrict__ Qb, bf16_t* __restrict__ Kb, bf16_t* __restrict__ Vtb) {
  __shared__ __align__(16) bf16_t Asm[2][64 * A_STRIDE];

  const int tid  = threadIdx.x;
  const int lane = tid & 31;
  const int wave = tid >> 5;
  const int lm = lane & 15, hi = lane >> 4;
  const int wm = wave & 1,  wn = wave >> 1;
  const int m0 = blockIdx.x * 64;
  const int n0 = blockIdx.y * 256 + wn * 64;

  // one 16B chunk of the 64x32 A tile per thread
  const int arow = tid >> 2;           // 0..63
  const int ack  = (tid & 3) * 8;      // k elem 0/8/16/24
  const bf16_t* aglob = Xb + (size_t)(m0 + arow) * DIM + ack;
  const unsigned alds[2] = { lds_addr(&Asm[0][arow * A_STRIDE + ack]),
                             lds_addr(&Asm[1][arow * A_STRIDE + ack]) };
  async_copy_b128(alds[0], aglob);     // prologue: k0 = 0 -> buf 0

  v8f acc[2][4];
#pragma unroll
  for (int mf = 0; mf < 2; ++mf)
#pragma unroll
    for (int c = 0; c < 4; ++c) acc[mf][c] = zero8();

  for (int k0 = 0; k0 < DIM; k0 += 32) {
    const int cur = (k0 >> 5) & 1;
    if (k0 + 32 < DIM) {
      async_copy_b128(alds[cur ^ 1], aglob + k0 + 32);
      asm volatile("s_wait_asynccnt 0x1" ::: "memory");  // cur tile done
    } else {
      asm volatile("s_wait_asynccnt 0x0" ::: "memory");
    }
    __syncthreads();

    v16bf af0 = ld_frag(&Asm[cur][(wm * 32 + lm)      * A_STRIDE + 8 * hi]);
    v16bf af1 = ld_frag(&Asm[cur][(wm * 32 + 16 + lm) * A_STRIDE + 8 * hi]);
#pragma unroll
    for (int c = 0; c < 4; ++c) {
      v16bf b = ld_frag(Wb + (size_t)(n0 + 16 * c + lm) * DIM + k0 + 8 * hi);
      acc[0][c] = wmma_bf16(af0, b, acc[0][c]);
      acc[1][c] = wmma_bf16(af1, b, acc[1][c]);
    }
    __syncthreads();   // before next iteration overwrites buf cur^1's pair
  }

#pragma unroll
  for (int c = 0; c < 4; ++c) {
    const int bo  = n0 + 16 * c;
    const int s   = bo / DIM;
    const int rem = bo - s * DIM;
    const int h   = rem >> 6;
    const int d   = (rem & 63) + lm;
#pragma unroll
    for (int mf = 0; mf < 2; ++mf)
#pragma unroll
      for (int r = 0; r < 8; ++r) {
        const int t = m0 + wm * 32 + mf * 16 + r + 8 * hi;
        const int b = t >> 10;
        const int n = t & 1023;
        const int bh = b * HEADS + h;
        const float v = acc[mf][c][r];
        if (s == 0)      Qb [((size_t)bh * SEQ + n) * HDIM + d] = (bf16_t)(v * SCALE_Q);
        else if (s == 1) Kb [((size_t)bh * SEQ + n) * HDIM + d] = (bf16_t)v;
        else             Vtb[((size_t)bh * HDIM + d) * SEQ + n] = (bf16_t)v;
      }
  }
}

// ---------------------------------------------------------------------------
// Flash attention: grid (96 heads, 16 q-tiles), 4 waves x 16 query rows.
// K/V 64-key tiles staged in LDS (async, double buffered) shared by all waves.
// Per chunk: 8 WMMA QK^T, online softmax, P relayout via LDS, 8 WMMA P.V.
// ---------------------------------------------------------------------------
__global__ __launch_bounds__(128) void attn_kernel(
    const bf16_t* __restrict__ Qb, const bf16_t* __restrict__ Kb,
    const bf16_t* __restrict__ Vtb, bf16_t* __restrict__ Ab) {
  __shared__ __align__(16) bf16_t Kt[2][64 * KV_STRIDE];
  __shared__ __align__(16) bf16_t Vt[2][64 * KV_STRIDE];
  __shared__ __align__(16) bf16_t Pt[4][16 * 64];

  const int tid  = threadIdx.x;
  const int lane = tid & 31;
  const int wave = tid >> 5;
  const int lm = lane & 15, hi = lane >> 4;
  const int bh = blockIdx.x;
  const int q0 = blockIdx.y * 64 + wave * 16;

  // Q fragments (head dim 64 -> two 32-chunks), loaded once
  const bf16_t* qrow = Qb + ((size_t)bh * SEQ + q0 + lm) * HDIM;
  const v16bf aq0 = ld_frag(qrow + 0  + 8 * hi);
  const v16bf aq1 = ld_frag(qrow + 32 + 8 * hi);

  // async stage of K (64 keys x 64 d) and V^T (64 d x 64 keys): 8 x 16B / thread
  auto issue_kv = [&](int kc, int b) {
#pragma unroll
    for (int j = 0; j < 4; ++j) {
      const int idx = tid * 4 + j;          // 0..511
      const int row = idx >> 3;             // 0..63
      const int ck  = (idx & 7) * 8;        // 0..56
      async_copy_b128(lds_addr(&Kt[b][row * KV_STRIDE + ck]),
                      Kb + ((size_t)bh * SEQ + kc + row) * HDIM + ck);
      async_copy_b128(lds_addr(&Vt[b][row * KV_STRIDE + ck]),
                      Vtb + ((size_t)bh * HDIM + row) * SEQ + kc + ck);
    }
  };
  issue_kv(0, 0);

  v8f oacc[4];
#pragma unroll
  for (int c = 0; c < 4; ++c) oacc[c] = zero8();
  float mrow[8], lrow[8];
#pragma unroll
  for (int r = 0; r < 8; ++r) { mrow[r] = -1e30f; lrow[r] = 0.f; }

  for (int kc = 0; kc < SEQ; kc += 64) {
    const int cur = (kc >> 6) & 1;
    if (kc + 64 < SEQ) {
      issue_kv(kc + 64, cur ^ 1);
      asm volatile("s_wait_asynccnt 0x8" ::: "memory");  // cur batch (8) done
    } else {
      asm volatile("s_wait_asynccnt 0x0" ::: "memory");
    }
    __syncthreads();

    // ---- S = Q K^T (16 x 64 tile) ----
    v8f sacc[4];
#pragma unroll
    for (int c = 0; c < 4; ++c) sacc[c] = zero8();
#pragma unroll
    for (int c = 0; c < 4; ++c) {
      const bf16_t* kb = &Kt[cur][(16 * c + lm) * KV_STRIDE + 8 * hi];
      sacc[c] = wmma_bf16(aq0, ld_frag(kb),      sacc[c]);
      sacc[c] = wmma_bf16(aq1, ld_frag(kb + 32), sacc[c]);
    }

    // ---- online softmax (row reductions across 16-lane halves) ----
#pragma unroll
    for (int r = 0; r < 8; ++r) {
      float cm = fmaxf(fmaxf(sacc[0][r], sacc[1][r]),
                       fmaxf(sacc[2][r], sacc[3][r]));
      cm = fmaxf(cm, __shfl_xor(cm, 1));
      cm = fmaxf(cm, __shfl_xor(cm, 2));
      cm = fmaxf(cm, __shfl_xor(cm, 4));
      cm = fmaxf(cm, __shfl_xor(cm, 8));
      const float mnew = fmaxf(mrow[r], cm);
      const float sc = __expf(mrow[r] - mnew);
      float rs = 0.f;
#pragma unroll
      for (int c = 0; c < 4; ++c) {
        float p = __expf(sacc[c][r] - mnew);
        sacc[c][r] = p;
        rs += p;
      }
      rs += __shfl_xor(rs, 1);
      rs += __shfl_xor(rs, 2);
      rs += __shfl_xor(rs, 4);
      rs += __shfl_xor(rs, 8);
      lrow[r] = lrow[r] * sc + rs;
      mrow[r] = mnew;
#pragma unroll
      for (int c = 0; c < 4; ++c) oacc[c][r] *= sc;
    }

    // ---- P: C-fragment -> row-major 16x64 in LDS -> A-fragment ----
    bf16_t* pw = &Pt[wave][0];
#pragma unroll
    for (int r = 0; r < 8; ++r)
#pragma unroll
      for (int c = 0; c < 4; ++c)
        pw[(r + 8 * hi) * 64 + 16 * c + lm] = (bf16_t)sacc[c][r];
    asm volatile("s_wait_dscnt 0" ::: "memory");  // per-wave tile; DS in-order

    const bf16_t* prd = pw + lm * 64;
    v16bf ap0 = ld_frag(prd + 0  + 8 * hi);
    v16bf ap1 = ld_frag(prd + 32 + 8 * hi);

    // ---- O += P . V (V pre-transposed) ----
#pragma unroll
    for (int c = 0; c < 4; ++c) {
      const bf16_t* vb = &Vt[cur][(16 * c + lm) * KV_STRIDE + 8 * hi];
      oacc[c] = wmma_bf16(ap0, ld_frag(vb),      oacc[c]);
      oacc[c] = wmma_bf16(ap1, ld_frag(vb + 32), oacc[c]);
    }
    __syncthreads();   // all waves done reading buf cur before it's re-filled
  }

  // ---- normalize, store bf16 [B,N,C] for the projection GEMM ----
  const int b = bh / HEADS;
  const int h = bh % HEADS;
#pragma unroll
  for (int c = 0; c < 4; ++c) {
    const int d = 16 * c + lm;
#pragma unroll
    for (int r = 0; r < 8; ++r) {
      const int n = q0 + r + 8 * hi;
      const float v = oacc[c][r] / lrow[r];
      Ab[((size_t)(b * SEQ + n)) * DIM + h * HDIM + d] = (bf16_t)v;
    }
  }
}

// ---------------------------------------------------------------------------
// Projection GEMM: out[t,o] = sum_c A[t,c] * w_proj[o,c] + bias[o]  (f32 out)
// Same tiling as QKV GEMM (64x256 per block, async-staged A tile).
// ---------------------------------------------------------------------------
__global__ __launch_bounds__(256) void proj_gemm_kernel(
    const bf16_t* __restrict__ Ab, const bf16_t* __restrict__ Wb,
    const float* __restrict__ bias, float* __restrict__ out) {
  __shared__ __align__(16) bf16_t Asm[2][64 * A_STRIDE];

  const int tid  = threadIdx.x;
  const int lane = tid & 31;
  const int wave = tid >> 5;
  const int lm = lane & 15, hi = lane >> 4;
  const int wm = wave & 1,  wn = wave >> 1;
  const int m0 = blockIdx.x * 64;
  const int n0 = blockIdx.y * 256 + wn * 64;

  const int arow = tid >> 2;
  const int ack  = (tid & 3) * 8;
  const bf16_t* aglob = Ab + (size_t)(m0 + arow) * DIM + ack;
  const unsigned alds[2] = { lds_addr(&Asm[0][arow * A_STRIDE + ack]),
                             lds_addr(&Asm[1][arow * A_STRIDE + ack]) };
  async_copy_b128(alds[0], aglob);

  v8f acc[2][4];
#pragma unroll
  for (int mf = 0; mf < 2; ++mf)
#pragma unroll
    for (int c = 0; c < 4; ++c) acc[mf][c] = zero8();

  for (int k0 = 0; k0 < DIM; k0 += 32) {
    const int cur = (k0 >> 5) & 1;
    if (k0 + 32 < DIM) {
      async_copy_b128(alds[cur ^ 1], aglob + k0 + 32);
      asm volatile("s_wait_asynccnt 0x1" ::: "memory");
    } else {
      asm volatile("s_wait_asynccnt 0x0" ::: "memory");
    }
    __syncthreads();

    v16bf af0 = ld_frag(&Asm[cur][(wm * 32 + lm)      * A_STRIDE + 8 * hi]);
    v16bf af1 = ld_frag(&Asm[cur][(wm * 32 + 16 + lm) * A_STRIDE + 8 * hi]);
#pragma unroll
    for (int c = 0; c < 4; ++c) {
      v16bf b = ld_frag(Wb + (size_t)(n0 + 16 * c + lm) * DIM + k0 + 8 * hi);
      acc[0][c] = wmma_bf16(af0, b, acc[0][c]);
      acc[1][c] = wmma_bf16(af1, b, acc[1][c]);
    }
    __syncthreads();
  }

#pragma unroll
  for (int c = 0; c < 4; ++c) {
    const int o = n0 + 16 * c + lm;
    const float bv = bias[o];
#pragma unroll
    for (int mf = 0; mf < 2; ++mf)
#pragma unroll
      for (int r = 0; r < 8; ++r) {
        const int t = m0 + wm * 32 + mf * 16 + r + 8 * hi;
        out[(size_t)t * DIM + o] = acc[mf][c][r] + bv;
      }
  }
}

// ---------------------------------------------------------------------------
// Host launch
// ---------------------------------------------------------------------------
extern "C" void kernel_launch(void* const* d_in, const int* in_sizes, int n_in,
                              void* d_out, int out_size, void* d_ws, size_t ws_size,
                              hipStream_t stream) {
  const float* x      = (const float*)d_in[0];  // [8,1024,768]
  const float* w_qkv  = (const float*)d_in[1];  // [2304,768]
  const float* w_proj = (const float*)d_in[2];  // [768,768]
  const float* b_proj = (const float*)d_in[3];  // [768]
  float* out = (float*)d_out;                   // [8,1024,768] f32

  char* ws = (char*)d_ws;
  const size_t SZ_X   = (size_t)NTOK * DIM * 2;
  const size_t SZ_WQ  = (size_t)THREEC * DIM * 2;
  const size_t SZ_WP  = (size_t)DIM * DIM * 2;
  const size_t SZ_HED = (size_t)8 * HEADS * SEQ * HDIM * 2;
  size_t off = 0;
  bf16_t* Xb    = (bf16_t*)(ws + off); off += SZ_X;
  bf16_t* Wqkvb = (bf16_t*)(ws + off); off += SZ_WQ;
  bf16_t* Wprojb= (bf16_t*)(ws + off); off += SZ_WP;
  bf16_t* Qb    = (bf16_t*)(ws + off); off += SZ_HED;
  bf16_t* Kb    = (bf16_t*)(ws + off); off += SZ_HED;
  bf16_t* Vtb   = (bf16_t*)(ws + off); off += SZ_HED;
  bf16_t* Ab    = (bf16_t*)(ws + off); off += SZ_X;
  (void)ws_size; (void)in_sizes; (void)n_in; (void)out_size;

  {
    int n4 = (NTOK * DIM) / 4;
    cvt_f32_bf16<<<(n4 + 255) / 256, 256, 0, stream>>>(x, Xb, n4);
    n4 = (THREEC * DIM) / 4;
    cvt_f32_bf16<<<(n4 + 255) / 256, 256, 0, stream>>>(w_qkv, Wqkvb, n4);
    n4 = (DIM * DIM) / 4;
    cvt_f32_bf16<<<(n4 + 255) / 256, 256, 0, stream>>>(w_proj, Wprojb, n4);
  }

  // QKV GEMM: 8192 x 2304 x 768
  qkv_gemm_kernel<<<dim3(NTOK / 64, THREEC / 256), 256, 0, stream>>>(
      Xb, Wqkvb, Qb, Kb, Vtb);

  // Attention: 96 (b,h) x 16 query tiles of 64 rows
  attn_kernel<<<dim3(8 * HEADS, SEQ / 64), 128, 0, stream>>>(Qb, Kb, Vtb, Ab);

  // Projection GEMM: 8192 x 768 x 768 + bias
  proj_gemm_kernel<<<dim3(NTOK / 64, DIM / 256), 256, 0, stream>>>(
      Ab, Wprojb, b_proj, out);
}
